// Renderer_54881092108848
// MI455X (gfx1250) — matrix operand order repaired and checked by reading, not compiled, not run on hardware
//
#include <hip/hip_runtime.h>

// 3DGS fused renderer for gfx1250 (MI455X), wave32 + V_WMMA_F32_16X16X32_F16
// + GLOBAL_LOAD_ASYNC_TO_LDS_B128 chunk staging (ASYNCcnt) + coalesced epilogue.
// H = W = 256, N_GAUSS = 1024, CF = 15 (+1 ones channel -> 16 = WMMA N dim).

#define Hh 256
#define Ww 256
#define NG 1024
#define CFEAT 15

typedef __attribute__((ext_vector_type(16))) _Float16 v16h;
typedef __attribute__((ext_vector_type(8)))  float    v8f;

// workspace layout (bytes)
#define WS_TZ     0        // 1024 f32  depth keys
#define WS_PRAW   4096     // 1024 x 8 f32  unsorted params {ux,uy,cA,cB,cC,op,0,0}
#define WS_PSORT  36864    // 1024 x 8 f32  depth-sorted params
#define WS_FEATT  69632    // 16 x 1024 f16 transposed sorted features (B operand)

// ---------------------------------------------------------------------------
// Phase 1: per-gaussian projection / 2D covariance inversion
// ---------------------------------------------------------------------------
__global__ void gs_geom(const float* __restrict__ xyzs,
                        const float* __restrict__ opac,
                        const float* __restrict__ cov,
                        const float* __restrict__ Km,
                        const float* __restrict__ Em,
                        float* __restrict__ tzbuf,
                        float* __restrict__ praw) {
  int i = blockIdx.x * blockDim.x + threadIdx.x;
  if (i >= NG) return;

  const float fx = Km[0], fy = Km[4], px = Km[2], py = Km[5];
  float R[3][3], tr[3];
#pragma unroll
  for (int r = 0; r < 3; ++r) {
#pragma unroll
    for (int c = 0; c < 3; ++c) R[r][c] = Em[r * 4 + c];
    tr[r] = Em[r * 4 + 3];
  }

  const float p0 = xyzs[i * 3 + 0], p1 = xyzs[i * 3 + 1], p2 = xyzs[i * 3 + 2];
  const float t0 = R[0][0] * p0 + R[0][1] * p1 + R[0][2] * p2 + tr[0];
  const float t1 = R[1][0] * p0 + R[1][1] * p1 + R[1][2] * p2 + tr[1];
  const float tz = R[2][0] * p0 + R[2][1] * p1 + R[2][2] * p2 + tr[2];

  const float limx = 1.3f * ((float)Ww / (2.0f * fx));
  const float limy = 1.3f * ((float)Hh / (2.0f * fy));
  const float txc = fminf(fmaxf(t0 / tz, -limx), limx) * tz;
  const float tyc = fminf(fmaxf(t1 / tz, -limy), limy) * tz;
  const float z2 = tz * tz;

  float J[2][3] = {{fx / tz, 0.0f, -fx * txc / z2},
                   {0.0f, fy / tz, -fy * tyc / z2}};
  float Tm[2][3];
#pragma unroll
  for (int r = 0; r < 2; ++r)
#pragma unroll
    for (int c = 0; c < 3; ++c)
      Tm[r][c] = J[r][0] * R[0][c] + J[r][1] * R[1][c] + J[r][2] * R[2][c];

  const float* Cm = cov + i * 9;
  float Mk[2][3];
#pragma unroll
  for (int r = 0; r < 2; ++r)
#pragma unroll
    for (int k = 0; k < 3; ++k)
      Mk[r][k] = Tm[r][0] * Cm[0 * 3 + k] + Tm[r][1] * Cm[1 * 3 + k] + Tm[r][2] * Cm[2 * 3 + k];

  const float c00 = Mk[0][0] * Tm[0][0] + Mk[0][1] * Tm[0][1] + Mk[0][2] * Tm[0][2] + 0.3f;
  const float c01 = Mk[0][0] * Tm[1][0] + Mk[0][1] * Tm[1][1] + Mk[0][2] * Tm[1][2];
  const float c11 = Mk[1][0] * Tm[1][0] + Mk[1][1] * Tm[1][1] + Mk[1][2] * Tm[1][2] + 0.3f;

  const float det = c00 * c11 - c01 * c01;
  const float inv = 1.0f / det;
  const float cA = c11 * inv, cB = -c01 * inv, cC = c00 * inv;
  const float ux = fx * t0 / tz + px - 0.5f;
  const float uy = fy * t1 / tz + py - 0.5f;
  const bool valid = (tz > 0.2f) && (det > 0.0f);
  const float op = valid ? opac[i] : 0.0f;

  tzbuf[i] = tz;
  float* o = praw + i * 8;
  o[0] = ux; o[1] = uy; o[2] = cA; o[3] = cB; o[4] = cC; o[5] = op; o[6] = 0.f; o[7] = 0.f;
}

// ---------------------------------------------------------------------------
// Phase 2: stable ascending depth sort (rank by counting), scatter params and
// build the transposed f16 feature matrix featT[16][NG] (WMMA B operand).
// ---------------------------------------------------------------------------
__global__ void gs_sort(const float* __restrict__ tzbuf,
                        const float* __restrict__ praw,
                        const float* __restrict__ feats,
                        float* __restrict__ psort,
                        _Float16* __restrict__ featT) {
  __shared__ float stz[NG];
  const int tid = threadIdx.x;
#pragma unroll
  for (int k = 0; k < NG / 256; ++k) stz[tid + k * 256] = tzbuf[tid + k * 256];
  __syncthreads();

#pragma unroll
  for (int k = 0; k < NG / 256; ++k) {
    const int i = tid + k * 256;
    const float ti = stz[i];
    int rank = 0;
    for (int j = 0; j < NG; ++j) {
      const float tj = stz[j];
      rank += (tj < ti) || (tj == ti && j < i);   // stable argsort semantics
    }
    const float4* src = (const float4*)(praw + i * 8);
    float4* dst = (float4*)(psort + rank * 8);
    dst[0] = src[0];
    dst[1] = src[1];
#pragma unroll
    for (int c = 0; c < CFEAT; ++c)
      featT[c * NG + rank] = (_Float16)feats[i * CFEAT + c];
    featT[CFEAT * NG + rank] = (_Float16)1.0f;    // ones channel -> weight sum
  }
}

// ---------------------------------------------------------------------------
// Phase 3: compositing. One wave32 per 16-pixel tile. Per 32-gaussian chunk:
// async-copy 1KB of sorted params global->LDS (ASYNCcnt), compute 16
// alphas/lane (A layout: lane m, half h holds K in {8h..8h+7, 16+8h..+7}),
// stitch the per-pixel transmittance prefix with two shfl_xor(16) exchanges,
// pack weights into the f16 A fragment, accumulate the 16x16 pixel-by-channel
// tile with v_wmma_f32_16x16x32_f16. Epilogue staged via LDS for coalesced
// contiguous stores (15-ch block is 240 consecutive floats per tile).
// ---------------------------------------------------------------------------
__global__ __launch_bounds__(256) void gs_comp(const float* __restrict__ psort,
                                               const _Float16* __restrict__ featT,
                                               float* __restrict__ out_feat,
                                               float* __restrict__ out_alpha) {
  __shared__ float4 sg[8][32][2];                 // per-wave staged chunk params (1KB/wave)
  __shared__ float  sout[8][16][16];              // per-wave D-tile staging

  const int lane = threadIdx.x & 31;
  const int wave = threadIdx.x >> 5;
  const int tile = blockIdx.x * 8 + wave;         // 4096 tiles of 16 pixels
  const int ty = tile >> 4;                       // pixel row
  const int tx0 = (tile & 15) << 4;               // first pixel column
  const int m = lane & 15;                        // pixel index M of A/D rows
  const int half = lane >> 4;                     // which K-half this lane owns
  const float pxx = (float)(tx0 + m);
  const float pxy = (float)ty;

  float Tcur = 1.0f;                              // running transmittance
  v8f acc = {};                                   // 16 pixels x 16 channels

  // per-lane LDS byte offsets for async staging (flat LDS addr low 32b = offset)
  const unsigned ldsbase = (unsigned)(uintptr_t)(&sg[wave][0][0]);
  const unsigned voff = (unsigned)lane * 16u;

  for (int chunk = 0; chunk < NG / 32; ++chunk) {
    // close WAR window: previous chunk's ds_load reads must land in VGPRs
    // before the async engine overwrites the staging buffer
    asm volatile("s_wait_dscnt 0" ::: "memory");

    // async copy: 32 lanes x 2 x 16B = 1KB of {ux,uy,cA,cB,cC,op,pad,pad}
    {
      const void* gbase = (const void*)(psort + (size_t)chunk * 32 * 8);
      asm volatile(
          "global_load_async_to_lds_b128 %0, %2, %3\n\t"
          "global_load_async_to_lds_b128 %1, %2, %3 offset:512\n\t"
          "s_wait_asynccnt 0"
          :: "v"(ldsbase + voff), "v"(ldsbase + voff + 512u),
             "v"(voff), "s"(gbase)
          : "memory");
    }

    if (chunk + 1 < NG / 32)
      __builtin_prefetch(featT + ((lane & 15) * NG + (chunk + 1) * 32 + half * 16), 0, 1);

    // B fragment: lane holds channel (lane&15), K = chunk*32 + {8h..8h+7, 16+8h..+7}
    v16h Bh;
#pragma unroll
    for (int s = 0; s < 16; ++s)
      Bh[s] = featT[(lane & 15) * NG + chunk * 32 + ((s < 8) ? (half * 8 + s) : (16 + half * 8 + (s - 8)))];

    // 16 alphas for this lane's K positions (no sequential dependency)
    float aj[16];
#pragma unroll
    for (int j = 0; j < 16; ++j) {
      const int kpos = (j < 8) ? (half * 8 + j) : (16 + half * 8 + (j - 8));
      const float* gp = (const float*)&sg[wave][kpos][0];
      const float dx = gp[0] - pxx;
      const float dy = gp[1] - pxy;
      const float power = -0.5f * (gp[2] * dx * dx + gp[4] * dy * dy) - gp[3] * dx * dy;
      const float gg = (power <= 0.0f) ? __expf(power) : 0.0f;
      const float al = fminf(0.99f, gp[5] * gg);
      aj[j] = (al < (1.0f / 255.0f)) ? 0.0f : al;
    }

    // cross-half transmittance stitch: per-8-group products of (1-alpha)
    float pAp = 1.0f, pBp = 1.0f;
#pragma unroll
    for (int j = 0; j < 8; ++j) pAp *= (1.0f - aj[j]);
#pragma unroll
    for (int j = 8; j < 16; ++j) pBp *= (1.0f - aj[j]);
    const float qA = __shfl_xor(pAp, 16, 32);     // partner half's group product
    const float qB = __shfl_xor(pBp, 16, 32);

    const float TA = half ? Tcur * qA : Tcur;               // start of K group A
    const float TB = Tcur * pAp * qA * (half ? qB : 1.0f);  // start of K group B

    // weights w = Tprev * alpha, packed into the A-operand f16 layout
    v16h Ah;
    float t = TA;
#pragma unroll
    for (int j = 0; j < 8; ++j) { Ah[j] = (_Float16)(t * aj[j]); t *= (1.0f - aj[j]); }
    t = TB;
#pragma unroll
    for (int j = 8; j < 16; ++j) { Ah[j] = (_Float16)(t * aj[j]); t *= (1.0f - aj[j]); }

    // product over all 32 K of (1-alpha) is pAp*qA*pBp*qB in both halves
    Tcur *= pAp * qA * pBp * qB;

    acc = __builtin_amdgcn_wmma_f32_16x16x32_f16(
        /*neg_a=*/false, Ah, /*neg_b=*/false, Bh,
        /*c_mod=*/(short)0, acc, /*reuse_a=*/false, /*reuse_b=*/false);
  }

  // ---- epilogue: stage D tile to LDS, then contiguous coalesced stores ----
  // D layout: VGPR r = pixel (half*8 + r), lane&15 = channel
  const int ch = lane & 15;
#pragma unroll
  for (int r = 0; r < 8; ++r)
    sout[wave][half * 8 + r][ch] = acc[r];
  asm volatile("s_wait_dscnt 0" ::: "memory");    // intra-wave LDS RAW

  // 15-channel block: 240 consecutive floats starting at (ty*W+tx0)*15
  const int fbase = (ty * Ww + tx0) * CFEAT;
#pragma unroll
  for (int j0 = 0; j0 < 240; j0 += 32) {
    const int j = j0 + lane;
    if (j < 240) {
      const int p = j / CFEAT;
      const int c = j - p * CFEAT;
      out_feat[fbase + j] = sout[wave][p][c];
    }
  }
  // ones channel -> weight-sum image (16 consecutive floats)
  if (lane < 16)
    out_alpha[ty * Ww + tx0 + lane] = sout[wave][lane][CFEAT];
}

extern "C" void kernel_launch(void* const* d_in, const int* in_sizes, int n_in,
                              void* d_out, int out_size, void* d_ws, size_t ws_size,
                              hipStream_t stream) {
  (void)in_sizes; (void)n_in; (void)out_size; (void)ws_size;
  const float* xyzs  = (const float*)d_in[0];
  const float* feats = (const float*)d_in[1];
  const float* opac  = (const float*)d_in[2];
  const float* cov   = (const float*)d_in[3];
  const float* Km    = (const float*)d_in[4];
  const float* Em    = (const float*)d_in[5];

  char* ws = (char*)d_ws;
  float*    tzb   = (float*)(ws + WS_TZ);
  float*    praw  = (float*)(ws + WS_PRAW);
  float*    psort = (float*)(ws + WS_PSORT);
  _Float16* featT = (_Float16*)(ws + WS_FEATT);

  float* out_feat  = (float*)d_out;
  float* out_alpha = out_feat + (size_t)Hh * Ww * CFEAT;

  gs_geom<<<4, 256, 0, stream>>>(xyzs, opac, cov, Km, Em, tzb, praw);
  gs_sort<<<1, 256, 0, stream>>>(tzb, praw, feats, psort, featT);
  gs_comp<<<512, 256, 0, stream>>>(psort, featT, out_feat, out_alpha);
}
